// SamplingMental_53051436040331
// MI455X (gfx1250) — matrix operand order repaired and checked by reading, not compiled, not run on hardware
//
#include <hip/hip_runtime.h>
#include <hip/hip_bf16.h>

// ---------------- CDNA5 WMMA / TDM types ----------------
typedef __attribute__((ext_vector_type(16))) __bf16 v16bf;
typedef __attribute__((ext_vector_type(8)))  float  v8f;
typedef unsigned int uint4v __attribute__((ext_vector_type(4)));
typedef int          int4v  __attribute__((ext_vector_type(4)));
typedef int          int8v  __attribute__((ext_vector_type(8)));

#define B_TOT     1024
#define L_EV      2048
#define L2_EV     4096
#define NGRID     100
#define HID       128
#define GATES     512
#define MTILE     16          // batches per LSTM block
#define NBLK_LSTM (B_TOT / MTILE)

// f32 -> bf16 (round-to-nearest-even)
static __device__ __forceinline__ unsigned short f2bf(float f) {
    unsigned int u = __float_as_uint(f);
    u += 0x7FFFu + ((u >> 16) & 1u);
    return (unsigned short)(u >> 16);
}

// time_emb: i=0..4 ; ang = t / 10000^(2i/5) ; even->sin, odd->cos
static __device__ __forceinline__ void time_emb5(float t, float e[5]) {
    e[0] = __sinf(t);
    e[1] = __cosf(t * (1.0f / 39.810717055349734f));
    e[2] = __sinf(t * (1.0f / 1584.8931924611136f));
    e[3] = __cosf(t * (1.0f / 63095.734448019316f));
    e[4] = __sinf(t * (1.0f / 2511886.4315095823f));
}

static __device__ __forceinline__ float sigf(float x) {
    return 1.0f / (1.0f + __expf(-x));
}

// B-matrix (32x16 bf16): lanes 0-15 hold K=0..7,16..23, lanes 16-31 hold K=8..15,24..31
static __device__ __forceinline__ int bkmap(int L, int e) {
    return e + ((L >= 16) ? 8 : 0) + ((e >= 8) ? 8 : 0);
}

// ---------------------------------------------------------------------
// Tensor Data Mover: 1-D row copy (n x f32, contiguous) global -> LDS.
// D# per CDNA5 ISA ch.8: group0 = {count/lds/global/type}, group1 = dims.
// ---------------------------------------------------------------------
static __device__ __forceinline__ void tdm_load_row(unsigned lds_off,
                                                    const float* gsrc, int n) {
    unsigned long long ga = (unsigned long long)(uintptr_t)gsrc;
    uint4v g0;
    g0[0] = 1u;                                        // count=1, user mode
    g0[1] = lds_off;                                   // lds_addr (bytes)
    g0[2] = (unsigned)(ga & 0xFFFFFFFFu);              // global_addr[31:0]
    g0[3] = (unsigned)((ga >> 32) & 0x01FFFFFFu)       // global_addr[56:32]
          | (2u << 30);                                // type = 2 ("image")
    int8v g1;
    g1[0] = (int)(2u << 16);                           // data_size = 4 bytes
    g1[1] = (int)(((unsigned)n & 0xFFFFu) << 16);      // tensor_dim0[15:0]
    g1[2] = (int)((((unsigned)n >> 16) & 0xFFFFu)      // tensor_dim0[31:16]
          | (1u << 16));                               // tensor_dim1 = 1
    g1[3] = (int)(((unsigned)n & 0xFFFFu) << 16);      // tile_dim0 = n
    g1[4] = 0;                                         // tile_dim1/2 unused
    g1[5] = n;                                         // tensor_dim0_stride
    g1[6] = 0;
    g1[7] = 0;
    int4v z4 = {0, 0, 0, 0};
#if __clang_major__ >= 23
    int8v z8 = {0, 0, 0, 0, 0, 0, 0, 0};
    __builtin_amdgcn_tensor_load_to_lds(g0, g1, z4, z4, z8, 0);
#else
    __builtin_amdgcn_tensor_load_to_lds(g0, g1, z4, z4, 0);
#endif
}

// =====================================================================
// Kernel 1: attention precompute.  One block per batch.
// att[(s*B + b)*8 + d]
// LDS: stage f32[4096] | emb f32[4096*5] | red f32[64]
// =====================================================================
__global__ void attn_kernel(const float* __restrict__ ta1,
                            const float* __restrict__ ta2,
                            float* __restrict__ att) {
    extern __shared__ char smem[];
    float* stage = (float*)smem;              // [4096] raw times via TDM
    float* emb   = stage + L2_EV;             // [4096][5]
    float* red   = emb + L2_EV * 5;           // [64] reduction scratch

    const int b    = blockIdx.x;
    const int tid  = threadIdx.x;
    const int wave = tid >> 5;
    const int lane = tid & 31;
    const float inv_scale = 0.35355339059327373f;  // 1/sqrt(8)

    // ---- TDM: DMA both time rows into LDS, wave 0 issues + waits ----
    if (tid < 32) {
        unsigned base = (unsigned)(uintptr_t)stage;
        tdm_load_row(base,            ta1 + (size_t)b * L_EV, L_EV);
        tdm_load_row(base + L_EV * 4, ta2 + (size_t)b * L_EV, L_EV);
        __builtin_amdgcn_s_wait_tensorcnt(0);
    }
    __syncthreads();

    // ---- embed all 4096 events into LDS ----
    #pragma unroll
    for (int k = 0; k < 16; ++k) {
        int l = tid + k * 256;
        float t = stage[l];
        float e[5]; time_emb5(t, e);
        #pragma unroll
        for (int d = 0; d < 5; ++d) emb[l * 5 + d] = e[d];
    }
    __syncthreads();

    for (int s = 0; s < NGRID; ++s) {
        float q[5]; time_emb5((float)(s + 1) * 0.1f, q);

        // pass 1: scores (registers) + max
        float scr[16];
        float mx = -1e30f;
        #pragma unroll
        for (int k = 0; k < 16; ++k) {
            int l = tid + k * 256;
            const float* e = &emb[l * 5];
            float sc = (q[0]*e[0] + q[1]*e[1] + q[2]*e[2] + q[3]*e[3] + q[4]*e[4]) * inv_scale;
            scr[k] = sc;
            mx = fmaxf(mx, sc);
        }
        #pragma unroll
        for (int o = 16; o > 0; o >>= 1) mx = fmaxf(mx, __shfl_xor(mx, o));
        if (lane == 0) red[wave] = mx;
        __syncthreads();
        #pragma unroll
        for (int w = 0; w < 8; ++w) mx = fmaxf(mx, red[w]);
        __syncthreads();

        // pass 2: exp-weighted sums
        float sum = 0.f, a0 = 0.f, a1 = 0.f, a2 = 0.f, a3 = 0.f, a4 = 0.f, m1 = 0.f;
        #pragma unroll
        for (int k = 0; k < 16; ++k) {
            int l = tid + k * 256;
            float w = __expf(scr[k] - mx);
            const float* e = &emb[l * 5];
            sum += w;
            a0 += w * e[0]; a1 += w * e[1]; a2 += w * e[2];
            a3 += w * e[3]; a4 += w * e[4];
            if (l < L_EV) m1 += w;
        }
        #pragma unroll
        for (int o = 16; o > 0; o >>= 1) {
            sum += __shfl_xor(sum, o);
            a0 += __shfl_xor(a0, o); a1 += __shfl_xor(a1, o);
            a2 += __shfl_xor(a2, o); a3 += __shfl_xor(a3, o);
            a4 += __shfl_xor(a4, o); m1 += __shfl_xor(m1, o);
        }
        if (lane == 0) {
            float* r = &red[wave * 7];
            r[0] = sum; r[1] = a0; r[2] = a1; r[3] = a2; r[4] = a3; r[5] = a4; r[6] = m1;
        }
        __syncthreads();
        if (tid == 0) {
            float S = 0, A0 = 0, A1 = 0, A2 = 0, A3 = 0, A4 = 0, M1 = 0;
            #pragma unroll
            for (int w = 0; w < 8; ++w) {
                const float* r = &red[w * 7];
                S += r[0]; A0 += r[1]; A1 += r[2]; A2 += r[3]; A3 += r[4]; A4 += r[5]; M1 += r[6];
            }
            float inv = 1.0f / S;
            float* o = &att[((size_t)s * B_TOT + b) * 8];
            o[0] = A0 * inv; o[1] = A1 * inv; o[2] = A2 * inv;
            o[3] = A3 * inv; o[4] = A4 * inv;
            o[5] = 0.0f;
            float mm = M1 * inv;
            o[6] = mm; o[7] = 1.0f - mm;
        }
        __syncthreads();
    }
}

// =====================================================================
// Kernel 2: LSTM recurrence with bf16 WMMA.  64 blocks x 256 threads.
// All weight B-fragments stream from LDS each step.  Accumulators are
// seeded with zero (rematerializable, never spilled); the bias is folded
// into the elementwise phase.  hx is written directly in bf16 A-fragment
// layout; the hxA RAW-hazard barrier sits right after the A-fragment
// loads so no WMMA accumulator is live across a barrier.
// Dynamic LDS layout (bytes):
//   whhB  ushort[65536]  @ 0        (B-frags: 4 kt x 32 tiles x 512)
//   wihB  ushort[16384]  @ 131072   (B-frags: 32 tiles x 512, K pad 21->32)
//   hxA   ushort[2048]   @ 163840   (A-frags: 4 kt x 512)
//   xA    ushort[512]    @ 167936   (A-frag : 1 x 512)
//   bias  float[512]     @ 168960
//   vvec  float[128]     @ 171008
//   hxf   float[2048]    @ 171520   (hx fp32 [16][128])
//   lt/dyn/dnext float[16] each, hev int[16], vc float  @ 179712..
// =====================================================================
#define LSTM_SMEM 180224

__global__ void lstm_kernel(const float* __restrict__ att,
                            const float* __restrict__ u,
                            const float* __restrict__ W_ih,
                            const float* __restrict__ W_hh,
                            const float* __restrict__ b_ih,
                            const float* __restrict__ b_hh,
                            const float* __restrict__ w1,
                            const float* __restrict__ b1,
                            const float* __restrict__ w2,
                            const float* __restrict__ b2,
                            float* __restrict__ out) {
    extern __shared__ char smem[];
    unsigned short* whhB = (unsigned short*)(smem);
    unsigned short* wihB = (unsigned short*)(smem + 131072);
    unsigned short* hxA  = (unsigned short*)(smem + 163840);
    unsigned short* xA   = (unsigned short*)(smem + 167936);
    float* bias  = (float*)(smem + 168960);
    float* vvec  = (float*)(smem + 171008);
    float* hxf   = (float*)(smem + 171520);
    float* lt    = (float*)(smem + 179712);
    float* dyn   = (float*)(smem + 179776);
    float* dnext = (float*)(smem + 179840);
    int*   hev   = (int*)  (smem + 179904);
    float* vc    = (float*)(smem + 179968);

    const int tid  = threadIdx.x;
    const int wave = tid >> 5;
    const int lane = tid & 31;

    // ---- init: swizzle weights into WMMA B-fragment order ----
    for (int idx = tid; idx < 65536; idx += 256) {
        int f = idx >> 9, r = idx & 511;
        int L = r >> 4, e = r & 15;
        int kt = f >> 5, t = f & 31;
        int nn = t * 16 + (L & 15);
        int kk = kt * 32 + bkmap(L, e);
        whhB[idx] = f2bf(W_hh[nn * HID + kk]);     // W_hh is [512][128] row-major
    }
    for (int idx = tid; idx < 16384; idx += 256) {
        int f = idx >> 9, r = idx & 511;
        int L = r >> 4, e = r & 15;
        int t = f;                                  // kt == 0
        int nn = t * 16 + (L & 15);
        int kk = bkmap(L, e);
        wihB[idx] = (kk < 21) ? f2bf(W_ih[nn * 21 + kk]) : (unsigned short)0;
    }
    for (int idx = tid; idx < GATES; idx += 256) bias[idx] = b_ih[idx] + b_hh[idx];
    for (int k = tid; k < HID; k += 256) {
        float s = 0.f;
        #pragma unroll
        for (int j = 0; j < 5; ++j) s += w2[j] * w1[j * HID + k];
        vvec[k] = s;
    }
    for (int idx = tid; idx < MTILE * HID; idx += 256) hxf[idx] = 0.f;
    for (int idx = tid; idx < 2048; idx += 256) hxA[idx] = 0;
    if (tid < MTILE) { lt[tid] = 0.f; dyn[tid] = 0.f; dnext[tid] = 0.f; hev[tid] = 0; }
    if (tid == 0) {
        float c = b2[0];
        #pragma unroll
        for (int j = 0; j < 5; ++j) c += w2[j] * b1[j];
        vc[0] = c;
    }
    __syncthreads();

    float bb_r[4];
    #pragma unroll
    for (int g = 0; g < 4; ++g)
        bb_r[g] = bias[(g * 8 + wave) * 16 + (lane & 15)];

    // hx A-fragment write-back mapping for this lane's hidden column
    const int hcol = wave * 16 + (lane & 15);
    const int kt_w = hcol >> 5;
    const int kp   = hcol & 31;

    float cx[8];
    #pragma unroll
    for (int r = 0; r < 8; ++r) cx[r] = 0.f;

    for (int s = 0; s < NGRID; ++s) {
        // ---- phase A: build x rows (threads 0..15, one per batch) ----
        if (tid < MTILE) {
            const int m  = tid;
            const int bg = blockIdx.x * MTILE + m;
            float xr[32];
            const float* at = &att[((size_t)s * B_TOT + bg) * 8];
            if (s + 1 < NGRID) __builtin_prefetch(&att[((size_t)(s + 1) * B_TOT + bg) * 8], 0, 0);
            #pragma unroll
            for (int d = 0; d < 8; ++d) xr[d] = at[d];
            float me[5]; time_emb5(lt[m], me);
            float he = hev[m] ? 1.0f : 0.0f;
            #pragma unroll
            for (int d = 0; d < 5; ++d) xr[8 + d] = me[d] * he;
            xr[13] = he; xr[14] = 0.f; xr[15] = 0.f;     // ONEHOT_M * has_event
            float dt = dyn[m] + 0.1f;
            dnext[m] = dt;
            float ce[5]; time_emb5(dt, ce);
            #pragma unroll
            for (int d = 0; d < 5; ++d) xr[16 + d] = ce[d];
            #pragma unroll
            for (int d = 21; d < 32; ++d) xr[d] = 0.f;
            // scatter row m into the A fragment
            #pragma unroll
            for (int k = 0; k < 32; ++k) {
                int L, e;
                if      (k <  8) { L = m;      e = k;      }
                else if (k < 16) { L = m + 16; e = k - 8;  }
                else if (k < 24) { L = m;      e = k - 8;  }
                else             { L = m + 16; e = k - 16; }
                xA[L * 16 + e] = f2bf(xr[k]);
            }
        }
        __syncthreads();

        // ---- phase C: load A fragments (only readers of xA / hxA) ----
        v16bf aih = *(const v16bf*)&xA[lane * 16];
        v16bf ahh[4];
        #pragma unroll
        for (int kt = 0; kt < 4; ++kt)
            ahh[kt] = *(const v16bf*)&hxA[kt * 512 + lane * 16];
        __syncthreads();   // all hxA reads done before phase D rewrites it

        // ---- GEMM: each wave -> 4 gate tiles of its 16 hidden cols ----
        v8f gacc[4];
        #pragma unroll
        for (int g = 0; g < 4; ++g) {
            v8f acc = {0.f, 0.f, 0.f, 0.f, 0.f, 0.f, 0.f, 0.f};
            v16bf bih = *(const v16bf*)&wihB[(g * 8 + wave) * 512 + lane * 16];
            acc = __builtin_amdgcn_wmma_f32_16x16x32_bf16(
                      false, aih, false, bih, (short)0, acc, false, false);
            #pragma unroll
            for (int kt = 0; kt < 4; ++kt) {
                v16bf bhh = *(const v16bf*)&whhB[(kt * 32 + g * 8 + wave) * 512 + lane * 16];
                acc = __builtin_amdgcn_wmma_f32_16x16x32_bf16(
                          false, ahh[kt], false, bhh, (short)0, acc, false, false);
            }
            gacc[g] = acc;
        }

        // ---- phase D: LSTM elementwise (bias folded in); wave owns
        //      hidden cols [wave*16, wave*16+16) ----
        #pragma unroll
        for (int r = 0; r < 8; ++r) {
            float i_ = gacc[0][r] + bb_r[0];
            float f_ = gacc[1][r] + bb_r[1];
            float g_ = gacc[2][r] + bb_r[2];
            float o_ = gacc[3][r] + bb_r[3];
            float c  = sigf(f_) * cx[r] + sigf(i_) * tanhf(g_);
            cx[r] = c;
            float h = sigf(o_) * tanhf(c);
            int m = r + ((lane >> 4) << 3);
            hxf[m * HID + hcol] = h;
            // write hx directly in bf16 A-fragment layout for next step
            int L, e;
            if      (kp <  8) { L = m;      e = kp;      }
            else if (kp < 16) { L = m + 16; e = kp - 8;  }
            else if (kp < 24) { L = m;      e = kp - 8;  }
            else              { L = m + 16; e = kp - 16; }
            hxA[kt_w * 512 + L * 16 + e] = f2bf(h);
        }
        __syncthreads();

        // ---- phase E: hz = sigmoid(hx . v + c), sample event, update state ----
        {
            int m = tid >> 4, p = tid & 15;
            float acc = 0.f;
            #pragma unroll
            for (int k = 0; k < 8; ++k)
                acc += hxf[m * HID + p * 8 + k] * vvec[p * 8 + k];
            #pragma unroll
            for (int o = 8; o > 0; o >>= 1) acc += __shfl_xor(acc, o, 16);
            if (p == 0) {
                float hz = sigf(acc + vc[0]);
                int bg = blockIdx.x * MTILE + m;
                float ut = u[(size_t)bg * NGRID + s];
                int ev = (ut < hz) ? 1 : 0;
                out[(size_t)s * B_TOT + bg] = hz;
                out[(size_t)NGRID * B_TOT + (size_t)s * B_TOT + bg] = ev ? 1.0f : 0.0f;
                float ct = (float)(s + 1) * 0.1f;
                lt[m]  = ev ? ct : lt[m];
                hev[m] = hev[m] | ev;
                dyn[m] = ev ? 0.0f : dnext[m];
            }
        }
        __syncthreads();
    }
}

// =====================================================================
extern "C" void kernel_launch(void* const* d_in, const int* in_sizes, int n_in,
                              void* d_out, int out_size, void* d_ws, size_t ws_size,
                              hipStream_t stream) {
    (void)in_sizes; (void)n_in; (void)out_size; (void)ws_size;
    const float* ta1  = (const float*)d_in[0];
    const float* ta2  = (const float*)d_in[1];
    const float* u    = (const float*)d_in[2];
    const float* W_ih = (const float*)d_in[3];
    const float* W_hh = (const float*)d_in[4];
    const float* b_ih = (const float*)d_in[5];
    const float* b_hh = (const float*)d_in[6];
    const float* w1   = (const float*)d_in[7];
    const float* b1   = (const float*)d_in[8];
    const float* w2   = (const float*)d_in[9];
    const float* b2   = (const float*)d_in[10];

    float* att = (float*)d_ws;   // [100][1024][8] f32 = 3.2 MB

    size_t smem_attn = (size_t)(L2_EV + L2_EV * 5 + 64) * sizeof(float);
    attn_kernel<<<B_TOT, 256, smem_attn, stream>>>(ta1, ta2, att);

    lstm_kernel<<<NBLK_LSTM, 256, LSTM_SMEM, stream>>>(
        att, u, W_ih, W_hh, b_ih, b_hh, w1, b1, w2, b2, (float*)d_out);
}
